// PerceiverAttention_21663815041179
// MI455X (gfx1250) — matrix operand order repaired and checked by reading, not compiled, not run on hardware
//
#include <hip/hip_runtime.h>

typedef __attribute__((ext_vector_type(16))) __bf16 v16bf;
typedef __attribute__((ext_vector_type(8)))  float  v8f;
typedef __attribute__((ext_vector_type(4)))  unsigned int u32x4;
typedef __attribute__((ext_vector_type(8)))  int          i32x8;
typedef __attribute__((ext_vector_type(4)))  int          i32x4;

union U16 { unsigned int u[8]; v16bf v; };

#if __has_builtin(__builtin_amdgcn_tensor_load_to_lds) && \
    __has_builtin(__builtin_amdgcn_s_wait_tensorcnt)
#define HAVE_TDM 1
#else
#define HAVE_TDM 0
#endif

__device__ __forceinline__ unsigned short f2bf(float f) {
  union { float f; unsigned int u; } c; c.f = f;
  unsigned int u = c.u;
  u += 0x7FFFu + ((u >> 16) & 1u);
  return (unsigned short)(u >> 16);
}
__device__ __forceinline__ float bf2f(unsigned short h) {
  union { unsigned int u; float f; } c; c.u = ((unsigned int)h) << 16;
  return c.f;
}

__device__ __forceinline__ unsigned lds_off(const void* p) {
  // generic 'shared' address: low 32 bits are the LDS byte offset
  return (unsigned)(unsigned long long)p;
}

#if HAVE_TDM
// 2D TDM tile load: tile0 elems (bf16, contiguous) x tile1 rows, row stride
// 'stride' elems. pad!=0 -> +1 DWORD of LDS padding every 32 DWORDs (i.e. a
// 64-elem bf16 row lands on a 66-halfword LDS pitch).
__device__ __forceinline__ void tdm_load_2d(unsigned lds, const void* gp,
                                            int tile0, int tile1, int stride,
                                            int pad) {
  unsigned long long ga = (unsigned long long)gp;
  unsigned galo = __builtin_amdgcn_readfirstlane((unsigned)ga);
  unsigned gahi = __builtin_amdgcn_readfirstlane((unsigned)(ga >> 32));
  unsigned ldsa = __builtin_amdgcn_readfirstlane(lds);
  unsigned strd = __builtin_amdgcn_readfirstlane((unsigned)stride);
  u32x4 g0;
  g0[0] = 1u;                                   // count=1, user descriptor
  g0[1] = ldsa;                                 // LDS byte address
  g0[2] = galo;                                 // global addr [31:0]
  g0[3] = (gahi & 0x01FFFFFFu) | 0x80000000u;   // global addr [56:32] | type=2
  int dw0 = 1 << 16;                            // data_size = 1 (2 bytes)
  if (pad) dw0 |= (1 << 20) | (4 << 22);        // pad_en, interval=32dw, amt=1dw
  const unsigned TD = 0x40000000u;              // huge tensor dims: no clipping
  i32x8 g1;
  g1[0] = dw0;
  g1[1] = (int)((TD & 0xFFFFu) << 16);                              // tdim0 lo
  g1[2] = (int)(((TD >> 16) & 0xFFFFu) | ((TD & 0xFFFFu) << 16));   // tdim0 hi|tdim1 lo
  g1[3] = (int)(((TD >> 16) & 0xFFFFu) | ((unsigned)tile0 << 16));  // tdim1 hi|tile0
  g1[4] = tile1;                                                    // tile1 | tile2=0
  g1[5] = (int)strd;                                                // dim0 stride lo
  g1[6] = 0;
  g1[7] = 0;
  i32x4 z4 = {0, 0, 0, 0};
#if __clang_major__ >= 23
  i32x8 z8 = {0, 0, 0, 0, 0, 0, 0, 0};
  __builtin_amdgcn_tensor_load_to_lds(g0, g1, z4, z4, z8, 0);
#else
  __builtin_amdgcn_tensor_load_to_lds(g0, g1, z4, z4, 0);
#endif
}
#endif

// ---------------------------------------------------------------------------
// Kernel 1: fused LayerNorm of src + ltn -> bf16 ctx [66048, 768]
// ---------------------------------------------------------------------------
__global__ __launch_bounds__(256) void ln_kernel(
    const float* __restrict__ src, const float* __restrict__ ltn,
    const float* __restrict__ gs, const float* __restrict__ bs,
    const float* __restrict__ gl, const float* __restrict__ bl,
    unsigned short* __restrict__ ctx) {
  const int row = blockIdx.x;
  const int tid = threadIdx.x;
  const int bt = row / 8256;
  const int i  = row % 8256;
  const float* x; const float* g; const float* b;
  if (i < 8192) { x = src + ((size_t)bt * 8192 + i) * 768; g = gs; b = bs; }
  else          { x = ltn + ((size_t)bt * 64 + (i - 8192)) * 768; g = gl; b = bl; }
  float v0 = x[tid], v1 = x[tid + 256], v2 = x[tid + 512];
  __shared__ float r1[256], r2[256];
  r1[tid] = v0 + v1 + v2;
  r2[tid] = v0*v0 + v1*v1 + v2*v2;
  __syncthreads();
  for (int off = 128; off > 0; off >>= 1) {
    if (tid < off) { r1[tid] += r1[tid+off]; r2[tid] += r2[tid+off]; }
    __syncthreads();
  }
  float mu  = r1[0] * (1.0f/768.0f);
  float var = r2[0] * (1.0f/768.0f) - mu*mu;
  float rs  = rsqrtf(var + 1e-5f);
  unsigned short* o = ctx + (size_t)row * 768;
  o[tid]       = f2bf((v0 - mu) * rs * g[tid]       + b[tid]);
  o[tid + 256] = f2bf((v1 - mu) * rs * g[tid + 256] + b[tid + 256]);
  o[tid + 512] = f2bf((v2 - mu) * rs * g[tid + 512] + b[tid + 512]);
}

// ---------------------------------------------------------------------------
// Kernel 2: convert Wq / Wkv / Wo fp32 -> bf16 (total 1572864, grid 6144x256)
// ---------------------------------------------------------------------------
__global__ __launch_bounds__(256) void wconv_kernel(
    const float* __restrict__ Wq, const float* __restrict__ Wkv,
    const float* __restrict__ Wo,
    unsigned short* __restrict__ wq, unsigned short* __restrict__ wkv,
    unsigned short* __restrict__ wo) {
  int idx = blockIdx.x * 256 + threadIdx.x;
  if (idx < 393216)            wq[idx]            = f2bf(Wq[idx]);
  else if (idx < 1179648)      wkv[idx - 393216]  = f2bf(Wkv[idx - 393216]);
  else                         wo[idx - 1179648]  = f2bf(Wo[idx - 1179648]);
}

// ---------------------------------------------------------------------------
// Kernel 3: bf16 WMMA GEMM  C[M,N] = A[M,K] * B[K,N]
// block tile 128x128, K-step 32; 8 waves (4m x 2n), each wave 2x4 16x16 tiles.
// Double-buffered pipeline: TDM stages A(t+1) async while B(t+1) is loaded to
// registers and the WMMAs of tile t run; one barrier per K-step.
// AMODE 0: A row = m ; AMODE 1 (Q proj): A row = (m>>6)*8256 + 8192 + (m&63)
// ---------------------------------------------------------------------------
template<int AMODE, typename OutT>
__global__ __launch_bounds__(256) void gemm_bf16(
    const unsigned short* __restrict__ A, const unsigned short* __restrict__ B,
    OutT* __restrict__ C, int M, int N, int K, int lda) {
  __shared__ unsigned short sA[2][128 * 32];   // A tiles row-major, 32 bf16/row
  __shared__ unsigned short sBT[2][128 * 34];  // B tiles transposed [n][k], +pad
  const int tid = threadIdx.x;
  const int m0 = blockIdx.y * 128;
  const int n0 = blockIdx.x * 128;
  const int w = tid >> 5, lane = tid & 31;
  const int wm = w & 3, wn = w >> 2;
  const int lr = lane & 15, kh = lane >> 4;

  v8f acc[2][4];
  #pragma unroll
  for (int mi = 0; mi < 2; ++mi)
    #pragma unroll
    for (int ni = 0; ni < 4; ++ni) acc[mi][ni] = {};

  auto issue_A = [&](int buf, int k0) {
#if HAVE_TDM
    if (w == 0) {
      if (AMODE == 0) {
        tdm_load_2d(lds_off(sA[buf]), A + (size_t)m0 * lda + k0, 32, 128, lda, 0);
      } else {
        long r0 = ((long)(m0 >> 6)) * 8256 + 8192;
        tdm_load_2d(lds_off(sA[buf]),           A + r0 * lda + k0,          32, 64, lda, 0);
        tdm_load_2d(lds_off(sA[buf]) + 64 * 64, A + (r0 + 8256) * lda + k0, 32, 64, lda, 0);
      }
    }
#else
    for (int c = tid; c < 512; c += 256) {
      int row = c >> 2, cc = c & 3;
      int gm = m0 + row;
      long arow = (AMODE == 0) ? (long)gm
                               : ((long)(gm >> 6) * 8256 + 8192 + (gm & 63));
      uint4 d = *(const uint4*)(A + arow * (long)lda + k0 + cc * 8);
      *(uint4*)&sA[buf][row * 32 + cc * 8] = d;
    }
#endif
  };
  auto load_B_regs = [&](int k0, uint4* r) {
    #pragma unroll
    for (int i = 0; i < 2; ++i) {
      int c = tid + i * 256;
      int kr = c >> 4, cc = c & 15;
      r[i] = *(const uint4*)(B + (size_t)(k0 + kr) * N + n0 + cc * 8);
    }
  };
  auto store_B_lds = [&](int buf, const uint4* r) {
    #pragma unroll
    for (int i = 0; i < 2; ++i) {
      int c = tid + i * 256;
      int kr = c >> 4, cc = c & 15;
      const unsigned short* ds = (const unsigned short*)&r[i];
      #pragma unroll
      for (int j = 0; j < 8; ++j) sBT[buf][(cc * 8 + j) * 34 + kr] = ds[j];
    }
  };

  // prologue: stage tile 0
  uint4 breg[2];
  issue_A(0, 0);
  load_B_regs(0, breg);
  store_B_lds(0, breg);
#if HAVE_TDM
  if (w == 0) __builtin_amdgcn_s_wait_tensorcnt((short)0);
#endif
  __syncthreads();

  for (int k0 = 0; k0 < K; k0 += 32) {
    const int cur = (k0 >> 5) & 1, nxt = cur ^ 1;
    const bool more = (k0 + 32) < K;
    if (more) {
      issue_A(nxt, k0 + 32);          // async DMA into alternate buffer
      load_B_regs(k0 + 32, breg);     // HBM loads overlap the WMMAs below
      if (k0 + 64 < K)                // deep prefetch of B(t+2) into caches
        __builtin_prefetch(B + (size_t)(k0 + 64 + (tid >> 4)) * N + n0 + (tid & 15) * 8, 0, 0);
    }

    const unsigned int* uA = (const unsigned int*)sA[cur];
    const unsigned int* uB = (const unsigned int*)sBT[cur];
    U16 a[2], bm;
    #pragma unroll
    for (int mi = 0; mi < 2; ++mi) {
      int m = wm * 32 + mi * 16 + lr;
      #pragma unroll
      for (int j = 0; j < 8; ++j)
        a[mi].u[j] = uA[m * 16 + j + 8 * kh];
    }
    #pragma unroll
    for (int ni = 0; ni < 4; ++ni) {
      int n = wn * 64 + ni * 16 + lr;
      #pragma unroll
      for (int j = 0; j < 8; ++j)
        bm.u[j] = uB[n * 17 + 8 * kh + j];
      #pragma unroll
      for (int mi = 0; mi < 2; ++mi)
        acc[mi][ni] = __builtin_amdgcn_wmma_f32_16x16x32_bf16(
            false, a[mi].v, false, bm.v, (short)0, acc[mi][ni], false, false);
    }

    if (more) {
      store_B_lds(nxt, breg);
#if HAVE_TDM
      if (w == 0) __builtin_amdgcn_s_wait_tensorcnt((short)0);
#endif
    }
    __syncthreads();
  }

  #pragma unroll
  for (int mi = 0; mi < 2; ++mi)
    #pragma unroll
    for (int ni = 0; ni < 4; ++ni) {
      int n = n0 + wn * 64 + ni * 16 + lr;
      #pragma unroll
      for (int r = 0; r < 8; ++r) {
        int m = m0 + wm * 32 + mi * 16 + r + 8 * kh;
        float v = acc[mi][ni][r];
        if constexpr (sizeof(OutT) == 2) C[(size_t)m * N + n] = (OutT)f2bf(v);
        else                             C[(size_t)m * N + n] = (OutT)v;
      }
    }
}

// ---------------------------------------------------------------------------
// Kernel 4: flash attention, one block per (bt, head); 64 queries, 8256 keys
// Q/K/V tiles staged by TDM with hardware row padding (66-halfword pitch).
// dhead^-0.5 folded into the softmax read.
// ---------------------------------------------------------------------------
__global__ __launch_bounds__(256) void attn_kernel(
    const unsigned short* __restrict__ qws,   // [8][64][512] bf16
    const unsigned short* __restrict__ kvws,  // [8][8256][1024] bf16 (k|v)
    unsigned short* __restrict__ ows) {       // [8][64][512] bf16
  __shared__ unsigned short sQ[64 * 66];
  __shared__ unsigned short sK[64 * 66];      // [key][dim] == Bt for S=QKt
  __shared__ unsigned short sV[64 * 66];      // [key][dim] row-major
  __shared__ float          sS[64 * 66];
  __shared__ unsigned short sP[64 * 66];
  __shared__ float sM[64], sL[64], sAl[64];

  const int tid = threadIdx.x;
  const int bt = blockIdx.x >> 3, h = blockIdx.x & 7;
  const int w = tid >> 5, lane = tid & 31;
  const int wm = w & 3, wn = w >> 2;
  const int lr = lane & 15, kh = lane >> 4;

  const unsigned short* qp = qws  + (size_t)bt * 64 * 512 + h * 64;
  const unsigned short* kp = kvws + (size_t)bt * 8256 * 1024 + h * 64;
  const unsigned short* vp = kp + 512;

#if HAVE_TDM
  if (w == 0) {
    tdm_load_2d(lds_off(sQ), qp, 64, 64, 512, 1);
    __builtin_amdgcn_s_wait_tensorcnt((short)0);
  }
#else
  for (int c = tid; c < 512; c += 256) {
    int row = c >> 3, cc = c & 7;
    uint4 d = *(const uint4*)(qp + (size_t)row * 512 + cc * 8);
    const unsigned int* du = (const unsigned int*)&d;
    unsigned int* dst = (unsigned int*)&sQ[row * 66 + cc * 8];
    #pragma unroll
    for (int j = 0; j < 4; ++j) dst[j] = du[j];
  }
#endif
  if (tid < 64) { sM[tid] = -3.0e38f; sL[tid] = 0.0f; }

  v8f o[2]; o[0] = {}; o[1] = {};

  const unsigned int* uQ = (const unsigned int*)sQ;
  const unsigned int* uK = (const unsigned int*)sK;
  const unsigned int* uP = (const unsigned int*)sP;
  __syncthreads();

  for (int j0 = 0; j0 < 8256; j0 += 64) {
#if HAVE_TDM
    if (w == 0) {
      tdm_load_2d(lds_off(sK), kp + (size_t)j0 * 1024, 64, 64, 1024, 1);
      tdm_load_2d(lds_off(sV), vp + (size_t)j0 * 1024, 64, 64, 1024, 1);
      __builtin_amdgcn_s_wait_tensorcnt((short)0);
    }
#else
    for (int c = tid; c < 512; c += 256) {
      int row = c >> 3, cc = c & 7;
      uint4 d = *(const uint4*)(kp + (size_t)(j0 + row) * 1024 + cc * 8);
      const unsigned int* du = (const unsigned int*)&d;
      unsigned int* dst = (unsigned int*)&sK[row * 66 + cc * 8];
      #pragma unroll
      for (int j = 0; j < 4; ++j) dst[j] = du[j];
    }
    for (int c = tid; c < 512; c += 256) {
      int row = c >> 3, cc = c & 7;
      uint4 d = *(const uint4*)(vp + (size_t)(j0 + row) * 1024 + cc * 8);
      const unsigned int* du = (const unsigned int*)&d;
      unsigned int* dst = (unsigned int*)&sV[row * 66 + cc * 8];
      #pragma unroll
      for (int j = 0; j < 4; ++j) dst[j] = du[j];
    }
#endif
    __syncthreads();

    // S = Q * Kt (each wave: 1 m-tile x 2 n-tiles, K=64 -> 2 wmma steps)
    #pragma unroll
    for (int ni = 0; ni < 2; ++ni) {
      v8f s = {};
      #pragma unroll
      for (int kk = 0; kk < 2; ++kk) {
        U16 a, bm;
        int m = wm * 16 + lr;
        #pragma unroll
        for (int j = 0; j < 8; ++j)
          a.u[j] = uQ[m * 33 + kk * 16 + j + 8 * kh];
        int n = wn * 32 + ni * 16 + lr;
        #pragma unroll
        for (int j = 0; j < 8; ++j)
          bm.u[j] = uK[n * 33 + kk * 16 + 8 * kh + j];
        s = __builtin_amdgcn_wmma_f32_16x16x32_bf16(
            false, a.v, false, bm.v, (short)0, s, false, false);
      }
      #pragma unroll
      for (int r = 0; r < 8; ++r)
        sS[(wm * 16 + r + 8 * kh) * 66 + wn * 32 + ni * 16 + lr] = s[r];
    }
    __syncthreads();

    // online softmax: 4 threads per row, 16 cols each; scale = 0.125
    {
      int row = tid >> 2, q4 = tid & 3;
      const float* srow = &sS[row * 66 + q4 * 16];
      float mx = -3.0e38f;
      #pragma unroll
      for (int c = 0; c < 16; ++c) mx = fmaxf(mx, srow[c] * 0.125f);
      mx = fmaxf(mx, __shfl_xor(mx, 1, 32));
      mx = fmaxf(mx, __shfl_xor(mx, 2, 32));
      float mprev = sM[row];
      float mnew  = fmaxf(mprev, mx);
      float ps = 0.0f;
      unsigned short* prow = &sP[row * 66 + q4 * 16];
      #pragma unroll
      for (int c = 0; c < 16; ++c) {
        float p = __expf(srow[c] * 0.125f - mnew);
        prow[c] = f2bf(p);
        ps += p;
      }
      ps += __shfl_xor(ps, 1, 32);
      ps += __shfl_xor(ps, 2, 32);
      if (q4 == 0) {
        float al = __expf(mprev - mnew);
        sAl[row] = al;
        sL[row]  = sL[row] * al + ps;
        sM[row]  = mnew;
      }
    }
    __syncthreads();

    // O = O*alpha + P * V  (V row-major -> gather B fragment as b16 pairs)
    #pragma unroll
    for (int ni = 0; ni < 2; ++ni) {
      #pragma unroll
      for (int r = 0; r < 8; ++r) o[ni][r] *= sAl[wm * 16 + r + 8 * kh];
      int n = wn * 32 + ni * 16 + lr;
      #pragma unroll
      for (int kk = 0; kk < 2; ++kk) {
        U16 a, bm;
        int m = wm * 16 + lr;
        #pragma unroll
        for (int j = 0; j < 8; ++j)
          a.u[j] = uP[m * 33 + kk * 16 + j + 8 * kh];
        #pragma unroll
        for (int j = 0; j < 8; ++j) {
          int kb = kk * 32 + 16 * kh + 2 * j;
          unsigned lo = sV[kb * 66 + n];
          unsigned hi = sV[(kb + 1) * 66 + n];
          bm.u[j] = lo | (hi << 16);
        }
        o[ni] = __builtin_amdgcn_wmma_f32_16x16x32_bf16(
            false, a.v, false, bm.v, (short)0, o[ni], false, false);
      }
    }
    __syncthreads();
  }

  // normalize and store
  #pragma unroll
  for (int ni = 0; ni < 2; ++ni)
    #pragma unroll
    for (int r = 0; r < 8; ++r) {
      int row = wm * 16 + r + 8 * kh;
      int col = wn * 32 + ni * 16 + lr;
      float v = o[ni][r] / sL[row];
      ows[((size_t)bt * 64 + row) * 512 + h * 64 + col] = f2bf(v);
    }
}

// ---------------------------------------------------------------------------
// launcher
// ---------------------------------------------------------------------------
extern "C" void kernel_launch(void* const* d_in, const int* in_sizes, int n_in,
                              void* d_out, int out_size, void* d_ws, size_t ws_size,
                              hipStream_t stream) {
  (void)in_sizes; (void)n_in; (void)out_size; (void)ws_size;
  const float* src = (const float*)d_in[0];
  const float* ltn = (const float*)d_in[1];
  const float* gs  = (const float*)d_in[2];
  const float* bs  = (const float*)d_in[3];
  const float* gl  = (const float*)d_in[4];
  const float* bl  = (const float*)d_in[5];
  const float* Wq  = (const float*)d_in[6];
  const float* Wkv = (const float*)d_in[7];
  const float* Wo  = (const float*)d_in[8];

  char* ws = (char*)d_ws;
  unsigned short* ctx = (unsigned short*)(ws);              // 66048*768 bf16
  unsigned short* wq  = (unsigned short*)(ws + 101449728);  // 768*512
  unsigned short* wkv = (unsigned short*)(ws + 102236160);  // 768*1024
  unsigned short* wo  = (unsigned short*)(ws + 103809024);  // 512*768
  unsigned short* qb  = (unsigned short*)(ws + 104595456);  // 512*512
  unsigned short* kvb = (unsigned short*)(ws + 105119744);  // 66048*1024
  unsigned short* ab  = (unsigned short*)(ws + 240386048);  // 512*512

  ln_kernel<<<66048, 256, 0, stream>>>(src, ltn, gs, bs, gl, bl, ctx);
  wconv_kernel<<<6144, 256, 0, stream>>>(Wq, Wkv, Wo, wq, wkv, wo);
  // kv = ctx @ Wkv   [66048,1024]
  gemm_bf16<0, unsigned short><<<dim3(8, 516), 256, 0, stream>>>(
      ctx, wkv, kvb, 66048, 1024, 768, 768);
  // q = ltn_n @ Wq   [512,512]  (A rows strided into ctx)
  gemm_bf16<1, unsigned short><<<dim3(4, 4), 256, 0, stream>>>(
      ctx, wq, qb, 512, 512, 768, 768);
  attn_kernel<<<64, 256, 0, stream>>>(qb, kvb, ab);
  // out = attn @ Wo  [512,768] fp32 -> d_out
  gemm_bf16<0, float><<<dim3(6, 4), 256, 0, stream>>>(
      ab, wo, (float*)d_out, 512, 768, 512, 512);
}
